// GptOssAttention_90778428768674
// MI455X (gfx1250) — matrix operand order repaired and checked by reading, not compiled, not run on hardware
//
#include <hip/hip_runtime.h>
#include <cstdint>
#include <cstddef>

// Problem constants (from reference)
#define TT      1024
#define HID     2880
#define NH      64
#define NKV     8
#define HDIM    64
#define QD      (NH * HDIM)    // 4096
#define KD      (NKV * HDIM)   // 512
#define WIN     128

// GEMM workgroup tiling
#define GBM 128
#define GBN 64
#define GBK 32

typedef __attribute__((ext_vector_type(16))) __bf16 v16bf;
typedef __attribute__((ext_vector_type(8)))  float  v8f;

union U8 { unsigned int u[8]; v16bf v; };
union F8 { float f[8]; v8f v; };

// ----- feature probes (compile-safe: guarded, with fallbacks) --------------
#if defined(__has_builtin)
#  if __has_builtin(__builtin_amdgcn_tensor_load_to_lds)
#    define HAVE_TDM 1
#  endif
#  if __has_builtin(__builtin_amdgcn_global_load_async_to_lds_b128)
#    define HAVE_ASYNC_LDS 1
#  endif
#endif
#ifndef HAVE_TDM
#  define HAVE_TDM 0
#endif
#ifndef HAVE_ASYNC_LDS
#  define HAVE_ASYNC_LDS 0
#endif

static __device__ __forceinline__ unsigned short f2bf(float x) {
  unsigned int u = __builtin_bit_cast(unsigned int, x);
  u += 0x7FFFu + ((u >> 16) & 1u);          // round-to-nearest-even
  return (unsigned short)(u >> 16);
}

static __device__ __forceinline__ v8f wmma_bf16(const U8& a, const U8& b, v8f c) {
  // D(16x16 f32) = A(16x32 bf16) * B(32x16 bf16) + C
  return __builtin_amdgcn_wmma_f32_16x16x32_bf16(
      false, a.v, false, b.v, (short)0, c, false, false);
}

#if HAVE_TDM
typedef __attribute__((ext_vector_type(4))) unsigned int u32x4;
typedef __attribute__((ext_vector_type(8))) int i32x8;
typedef __attribute__((ext_vector_type(4))) int i32x4;

// Build a 2D TDM descriptor (rows x cols, elements are 2 bytes, row stride in
// elements) and issue TENSOR_LOAD_TO_LDS.  D# layout per CDNA5 ISA ch.8.
static __device__ __forceinline__ void tdm_load_2d(
    const unsigned short* lds_dst, const unsigned short* gsrc,
    int rows, int cols, int row_stride_elems) {
  unsigned long long ga = (unsigned long long)(size_t)gsrc;
  unsigned int lds = (unsigned int)(size_t)lds_dst;   // low 32b = LDS offset
  u32x4 g0;
  g0[0] = 1u;                                          // count=1, user mode
  g0[1] = lds;                                         // lds_addr
  g0[2] = (unsigned int)(ga & 0xFFFFFFFFu);            // global_addr[31:0]
  g0[3] = (unsigned int)((ga >> 32) & 0x01FFFFFFu)     // global_addr[56:32]
        | (2u << 30);                                  // type = 2 ("image")
  i32x8 g1;
  g1[0] = (1 << 16);                                   // data_size = 1 (2B)
  g1[1] = (cols & 0xFFFF) << 16;                       // tensor_dim0[15:0]
  g1[2] = (rows & 0xFFFF) << 16;                       // tensor_dim1[15:0]
  g1[3] = (cols & 0xFFFF) << 16;                       // tile_dim0
  g1[4] = (rows & 0xFFFF);                             // tile_dim1 (tile_dim2=0)
  g1[5] = row_stride_elems;                            // tensor_dim0_stride lo
  g1[6] = 0;
  g1[7] = 0;
  i32x4 gz = {0, 0, 0, 0};
#if __clang_major__ >= 23
  i32x8 gz8 = {0, 0, 0, 0, 0, 0, 0, 0};
  __builtin_amdgcn_tensor_load_to_lds(g0, g1, gz, gz, gz8, 0);
#else
  __builtin_amdgcn_tensor_load_to_lds(g0, g1, gz, gz, 0);
#endif
}

static __device__ __forceinline__ void wait_tensorcnt0() {
#if __has_builtin(__builtin_amdgcn_s_wait_tensorcnt)
  __builtin_amdgcn_s_wait_tensorcnt(0);
#else
  asm volatile("s_wait_tensorcnt 0x0" ::: "memory");
#endif
}
// Allow 2 in-flight TDM ops (the next tile's A+B) while guaranteeing the
// current tile's 2 loads completed (TDM from one wave completes in order).
static __device__ __forceinline__ void wait_tensorcnt2() {
#if __has_builtin(__builtin_amdgcn_s_wait_tensorcnt)
  __builtin_amdgcn_s_wait_tensorcnt(2);
#else
  asm volatile("s_wait_tensorcnt 0x2" ::: "memory");
#endif
}
#endif  // HAVE_TDM

#if HAVE_ASYNC_LDS
// Toolchain prototype (from hipcc diagnostic): param0 = AS1 v4i*, param1 = AS3
// v4i*, then two immediate ints (offset, cpol).
typedef int gv4i __attribute__((vector_size(16)));
typedef __attribute__((address_space(1))) gv4i* gv4i_glb;
typedef __attribute__((address_space(3))) gv4i* gv4i_lds;

static __device__ __forceinline__ void cp_async_b128(void* lds, const void* g) {
  __builtin_amdgcn_global_load_async_to_lds_b128(
      (gv4i_glb)(size_t)g,
      (gv4i_lds)(unsigned)(size_t)lds,
      0, 0);
}
#endif

// ---------------------------------------------------------------------------
// f32 -> bf16 straight conversion
__global__ void k_cvt(const float* __restrict__ s, unsigned short* __restrict__ d, int n) {
  int i = blockIdx.x * blockDim.x + threadIdx.x;
  if (i < n) d[i] = f2bf(s[i]);
}

// f32 [R][C] -> bf16 [C][R] (transpose) : makes WMMA B-fragments K-contiguous
__global__ void k_cvtT(const float* __restrict__ s, unsigned short* __restrict__ d,
                       int R, int C) {
  int i = blockIdx.x * blockDim.x + threadIdx.x;
  if (i >= R * C) return;
  int r = i / C, c = i - r * C;
  d[(size_t)c * R + r] = f2bf(s[i]);
}

// ---------------------------------------------------------------------------
// LDS-tiled bf16 WMMA GEMM: C[M][N] f32 = A[M][K] * Bt[N][K]^T.
// 256 threads (8 waves) per block; block computes a 128x64 tile; wave w
// computes rows [w*16, w*16+16) x 64 cols (4 WMMA tiles, shared A fragment).
// TDM path: double-buffered LDS; wave 0 issues the NEXT tile's TDM loads and
// waits tensorcnt<=2 so the DMA of tile k+1 overlaps the WMMAs of tile k.
__global__ void __launch_bounds__(256) k_gemm_bf16(
    const unsigned short* __restrict__ A, const unsigned short* __restrict__ Bt,
    float* __restrict__ C, int M, int N, int K) {
#if HAVE_TDM
  __shared__ __align__(128) unsigned short Atile[2][GBM][GBK];  // 16 KB
  __shared__ __align__(128) unsigned short Btile[2][GBN][GBK];  //  8 KB
#else
  __shared__ __align__(128) unsigned short Atile[1][GBM][GBK];
  __shared__ __align__(128) unsigned short Btile[1][GBN][GBK];
#endif

  int nbn = N / GBN;
  int bm  = blockIdx.x / nbn;
  int bn  = blockIdx.x - bm * nbn;
  int m0  = bm * GBM, n0 = bn * GBN;
  if (m0 >= M) return;

  int tid  = threadIdx.x;
  int lane = tid & 31;
  int wv   = tid >> 5;              // wave in block: 0..7
  int lo = lane & 15, kh = lane >> 4;

  F8 acc[4];
#pragma unroll
  for (int nt = 0; nt < 4; ++nt)
#pragma unroll
    for (int r = 0; r < 8; ++r) acc[nt].f[r] = 0.0f;

#if HAVE_TDM
  // ---- double-buffered TDM pipeline -------------------------------------
  if (wv == 0) {
    tdm_load_2d(&Atile[0][0][0], A  + (size_t)m0 * K, GBM, GBK, K);
    tdm_load_2d(&Btile[0][0][0], Bt + (size_t)n0 * K, GBN, GBK, K);
  }
  int cur = 0;
  for (int kb = 0; kb < K; kb += GBK) {
    __syncthreads();                // all reads of buf[cur^1] done (WAR)
    if (wv == 0) {
      if (kb + GBK < K) {
        tdm_load_2d(&Atile[cur ^ 1][0][0], A  + (size_t)m0 * K + kb + GBK, GBM, GBK, K);
        tdm_load_2d(&Btile[cur ^ 1][0][0], Bt + (size_t)n0 * K + kb + GBK, GBN, GBK, K);
        wait_tensorcnt2();          // current tile landed; next tile in flight
      } else {
        wait_tensorcnt0();
      }
    }
    __syncthreads();                // release readers of buf[cur]

    U8 a;
#pragma unroll
    for (int i = 0; i < 8; ++i) {
      int ka = 16 * kh + 2 * i;
      a.u[i] = *(const unsigned int*)&Atile[cur][wv * 16 + lo][ka];
    }
#pragma unroll
    for (int nt = 0; nt < 4; ++nt) {
      U8 b;
#pragma unroll
      for (int i = 0; i < 8; ++i)
        b.u[i] = *(const unsigned int*)&Btile[cur][nt * 16 + lo][16 * kh + 2 * i];
      acc[nt].v = wmma_bf16(a, b, acc[nt].v);
    }
    cur ^= 1;
  }
#else
  // ---- single-buffered fallback (async-to-LDS or plain copies) ----------
  for (int kb = 0; kb < K; kb += GBK) {
    __syncthreads();                // protect LDS against WAR from prev iter
#if HAVE_ASYNC_LDS
#pragma unroll
    for (int j = 0; j < 2; ++j) {   // A: 512 b128 transfers
      int t2 = tid + 256 * j;
      int r = t2 >> 2, q = t2 & 3;
      cp_async_b128(&Atile[0][r][q * 8], A + (size_t)(m0 + r) * K + kb + q * 8);
    }
    {                               // B: 256 b128 transfers
      int r = tid >> 2, q = tid & 3;
      cp_async_b128(&Btile[0][r][q * 8], Bt + (size_t)(n0 + r) * K + kb + q * 8);
    }
    asm volatile("s_wait_asynccnt 0x0" ::: "memory");
#else
#pragma unroll
    for (int j = 0; j < 2; ++j) {   // A tile: 2x b128 per thread
      int t2 = tid + 256 * j;
      int r = t2 >> 2, q = t2 & 3;
      const unsigned short* src = A + (size_t)(m0 + r) * K + kb + q * 8;
      *(uint4*)&Atile[0][r][q * 8] = *(const uint4*)src;
      if (kb + GBK < K) __builtin_prefetch(src + GBK, 0, 1);
    }
    {                               // B tile: 1x b128 per thread
      int r = tid >> 2, q = tid & 3;
      const unsigned short* src = Bt + (size_t)(n0 + r) * K + kb + q * 8;
      *(uint4*)&Btile[0][r][q * 8] = *(const uint4*)src;
      if (kb + GBK < K) __builtin_prefetch(src + GBK, 0, 1);
    }
#endif
    __syncthreads();

    U8 a;
#pragma unroll
    for (int i = 0; i < 8; ++i) {
      int ka = 16 * kh + 2 * i;
      a.u[i] = *(const unsigned int*)&Atile[0][wv * 16 + lo][ka];
    }
#pragma unroll
    for (int nt = 0; nt < 4; ++nt) {
      U8 b;
#pragma unroll
      for (int i = 0; i < 8; ++i)
        b.u[i] = *(const unsigned int*)&Btile[0][nt * 16 + lo][16 * kh + 2 * i];
      acc[nt].v = wmma_bf16(a, b, acc[nt].v);
    }
  }
#endif

  // C layout: vgpr r, lanes0-15 -> M=r, lanes16-31 -> M=8+r ; N = lane&15
#pragma unroll
  for (int nt = 0; nt < 4; ++nt)
#pragma unroll
    for (int r = 0; r < 8; ++r)
      C[(size_t)(m0 + wv * 16 + kh * 8 + r) * N + n0 + nt * 16 + lo] = acc[nt].f[r];
}

// ---------------------------------------------------------------------------
// NeoX RoPE on Q and K (f32 in, bf16 out). One thread per (t, head, i<32).
__global__ void k_rope(const float* __restrict__ Qf, const float* __restrict__ Kf,
                       const int* __restrict__ pos,
                       unsigned short* __restrict__ Qbf, unsigned short* __restrict__ Kbf) {
  const int half = HDIM / 2;
  int i = blockIdx.x * blockDim.x + threadIdx.x;
  int total_q = TT * NH * half;
  int total   = total_q + TT * NKV * half;
  if (i >= total) return;
  bool isq = (i < total_q);
  int heads = isq ? NH : NKV;
  int j = isq ? i : i - total_q;
  int d   = j % half;
  int rem = j / half;
  int hh  = rem % heads;
  int t   = rem / heads;

  float p = (float)pos[t];
  // inv_freq = theta^(-d/half) = exp(-d * ln(150000)/32)
  float f = p * __expf(-(float)d * (11.918390573078392f / 32.0f));
  float sn, cs;
  __sincosf(f, &sn, &cs);

  const float* src = isq ? Qf : Kf;
  unsigned short* dst = isq ? Qbf : Kbf;
  size_t base = ((size_t)t * heads + hh) * HDIM;
  float x1 = src[base + d];
  float x2 = src[base + half + d];
  dst[base + d]        = f2bf(x1 * cs - x2 * sn);
  dst[base + half + d] = f2bf(x2 * cs + x1 * sn);
}

// ---------------------------------------------------------------------------
// Sliding-window attention with sinks, flash-style online softmax.
// One wave per (head, 16-query tile). Block = 4 waves.
__global__ void __launch_bounds__(128) k_attn(
    const unsigned short* __restrict__ Qbf,  // [T][H][HD] bf16, RoPE'd
    const unsigned short* __restrict__ Kbf,  // [T][KV][HD] bf16, RoPE'd
    const unsigned short* __restrict__ Vt,   // [KV*HD][T]  bf16 (transposed)
    const float* __restrict__ sinks,         // [H]
    unsigned short* __restrict__ Obf) {      // [T][H*HD] bf16
  __shared__ __align__(16) unsigned short pbuf[4][16][32];

  int wl   = threadIdx.x >> 5;
  int wave = blockIdx.x * 4 + wl;
  int lane = threadIdx.x & 31;
  int qt = wave & 63;
  int h  = wave >> 6;
  if (h >= NH) return;
  int q0  = qt * 16;
  int kvh = h >> 3;                // H/KV = 8 queries per kv head
  int lo = lane & 15, kh = lane >> 4;

  const float NEG = -3.0e38f;

  // Q A-fragments for K-dim 0..31 and 32..63
  U8 aq[2];
  const unsigned short* qrow = Qbf + ((size_t)(q0 + lo) * NH + h) * HDIM;
#pragma unroll
  for (int fgl = 0; fgl < 2; ++fgl)
#pragma unroll
    for (int i = 0; i < 8; ++i) {
      int k = 32 * fgl + 16 * kh + 2 * i;
      aq[fgl].u[i] = *(const unsigned int*)(qrow + k);
    }

  float mrow[8], lrow[8];
  F8 acc[4];
#pragma unroll
  for (int r = 0; r < 8; ++r) { mrow[r] = -1.0e30f; lrow[r] = 0.0f; }
#pragma unroll
  for (int nt = 0; nt < 4; ++nt)
#pragma unroll
    for (int r = 0; r < 8; ++r) acc[nt].f[r] = 0.0f;

  int s_begin = q0 - (WIN - 16);
  if (s_begin < 0) s_begin = 0;
  s_begin &= ~31;

  for (int sb = s_begin; sb <= q0 + 15; sb += 32) {
    // ---- logits: two 16x16 subtiles over this 32-key chunk
    F8 cl[2];
#pragma unroll
    for (int st = 0; st < 2; ++st) {
      int s = sb + 16 * st + lo;
      const unsigned short* krow = Kbf + ((size_t)s * NKV + kvh) * HDIM;
      U8 b0, b1;
#pragma unroll
      for (int i = 0; i < 8; ++i) {
        int kd = 16 * kh + 2 * i;
        b0.u[i] = *(const unsigned int*)(krow + kd);
        b1.u[i] = *(const unsigned int*)(krow + kd + 32);
      }
      v8f z = {};
      z = wmma_bf16(aq[0], b0, z);
      cl[st].v = wmma_bf16(aq[1], b1, z);
    }
    // ---- scale + mask
#pragma unroll
    for (int st = 0; st < 2; ++st)
#pragma unroll
      for (int r = 0; r < 8; ++r) {
        int t = q0 + kh * 8 + r;
        int s = sb + 16 * st + lo;
        float lg = cl[st].f[r] * 0.125f;                 // HD^-0.5
        bool ok = (s <= t) && (t - s < WIN);
        cl[st].f[r] = ok ? lg : NEG;
      }
    // ---- row max over the 32 keys (reduce within 16-lane half-wave)
    float rmax[8];
#pragma unroll
    for (int r = 0; r < 8; ++r) rmax[r] = fmaxf(cl[0].f[r], cl[1].f[r]);
#pragma unroll
    for (int dmask = 1; dmask < 16; dmask <<= 1)
#pragma unroll
      for (int r = 0; r < 8; ++r)
        rmax[r] = fmaxf(rmax[r], __shfl_xor(rmax[r], dmask, 32));

    float fac[8];
#pragma unroll
    for (int r = 0; r < 8; ++r) {
      float nm = fmaxf(mrow[r], rmax[r]);
      fac[r] = __expf(mrow[r] - nm);
      mrow[r] = nm;
    }
    // ---- p = exp(logit - m)
#pragma unroll
    for (int st = 0; st < 2; ++st)
#pragma unroll
      for (int r = 0; r < 8; ++r)
        cl[st].f[r] = __expf(cl[st].f[r] - mrow[r]);
    // ---- rescale running state; accumulate row sums
    float rs[8];
#pragma unroll
    for (int r = 0; r < 8; ++r) rs[r] = cl[0].f[r] + cl[1].f[r];
#pragma unroll
    for (int dmask = 1; dmask < 16; dmask <<= 1)
#pragma unroll
      for (int r = 0; r < 8; ++r) rs[r] += __shfl_xor(rs[r], dmask, 32);
#pragma unroll
    for (int r = 0; r < 8; ++r) lrow[r] = lrow[r] * fac[r] + rs[r];
#pragma unroll
    for (int nt = 0; nt < 4; ++nt)
#pragma unroll
      for (int r = 0; r < 8; ++r) acc[nt].f[r] *= fac[r];

    // ---- transpose P (C-layout) -> A-fragment layout through LDS
#pragma unroll
    for (int st = 0; st < 2; ++st)
#pragma unroll
      for (int r = 0; r < 8; ++r)
        pbuf[wl][kh * 8 + r][16 * st + lo] = f2bf(cl[st].f[r]);
    asm volatile("s_wait_dscnt 0" ::: "memory");   // cross-lane LDS RAW

    U8 pa;
#pragma unroll
    for (int i = 0; i < 8; ++i) {
      int k = 16 * kh + 2 * i;
      pa.u[i] = *(const unsigned int*)&pbuf[wl][lo][k];
    }

    // ---- P(16x32) @ V(32x64) in 4 column tiles
#pragma unroll
    for (int nt = 0; nt < 4; ++nt) {
      U8 bv;
#pragma unroll
      for (int i = 0; i < 8; ++i) {
        int s = sb + 16 * kh + 2 * i;
        int n = 16 * nt + lo;
        bv.u[i] = *(const unsigned int*)(Vt + (size_t)(kvh * HDIM + n) * TT + s);
      }
      acc[nt].v = wmma_bf16(pa, bv, acc[nt].v);
    }
    asm volatile("s_wait_dscnt 0" ::: "memory");   // keep LDS WAR ordered
  }

  // ---- fold in sink logit: mx = max(m, sink); denom += exp(sink - mx)
  float sk = sinks[h];
#pragma unroll
  for (int r = 0; r < 8; ++r) {
    float nm = fmaxf(mrow[r], sk);
    float fc = __expf(mrow[r] - nm);
    lrow[r] = lrow[r] * fc + __expf(sk - nm);
#pragma unroll
    for (int nt = 0; nt < 4; ++nt) acc[nt].f[r] *= fc;
  }

  // ---- write bf16 output [t][h*HD + hd]
#pragma unroll
  for (int nt = 0; nt < 4; ++nt)
#pragma unroll
    for (int r = 0; r < 8; ++r) {
      int t = q0 + kh * 8 + r;
      int n = 16 * nt + lo;
      Obf[((size_t)t * NH + h) * HDIM + n] = f2bf(acc[nt].f[r] / lrow[r]);
    }
}

// ---------------------------------------------------------------------------
extern "C" void kernel_launch(void* const* d_in, const int* in_sizes, int n_in,
                              void* d_out, int out_size, void* d_ws, size_t ws_size,
                              hipStream_t stream) {
  const float* hidden = (const float*)d_in[0];
  const int*   pos    = (const int*)d_in[1];
  const float* wq     = (const float*)d_in[2];
  const float* wk     = (const float*)d_in[3];
  const float* wv     = (const float*)d_in[4];
  const float* wo     = (const float*)d_in[5];
  const float* sinks  = (const float*)d_in[6];
  float* out = (float*)d_out;

  char* ws = (char*)d_ws;
  size_t off = 0;
  auto alloc = [&](size_t bytes) -> char* {
    char* p = ws + off;
    off += (bytes + 255) & ~(size_t)255;
    return p;
  };
  unsigned short* Xbf  = (unsigned short*)alloc((size_t)TT * HID * 2);
  unsigned short* Wqt  = (unsigned short*)alloc((size_t)QD * HID * 2);
  unsigned short* Wkt  = (unsigned short*)alloc((size_t)KD * HID * 2);
  unsigned short* Wvt  = (unsigned short*)alloc((size_t)KD * HID * 2);
  unsigned short* Wot  = (unsigned short*)alloc((size_t)HID * QD * 2);
  float*          Qf   = (float*)alloc((size_t)TT * QD * 4);
  float*          Kf   = (float*)alloc((size_t)TT * KD * 4);
  float*          Vf   = (float*)alloc((size_t)TT * KD * 4);
  unsigned short* Qbf  = (unsigned short*)alloc((size_t)TT * QD * 2);
  unsigned short* Kbf  = (unsigned short*)alloc((size_t)TT * KD * 2);
  unsigned short* Vtb  = (unsigned short*)alloc((size_t)KD * TT * 2);
  unsigned short* Attn = (unsigned short*)alloc((size_t)TT * QD * 2);

  auto cdiv = [](int a, int b) { return (a + b - 1) / b; };

  // 1) bf16 conversions (weights transposed so B-fragments are K-contiguous)
  k_cvt<<<cdiv(TT * HID, 256), 256, 0, stream>>>(hidden, Xbf, TT * HID);
  k_cvtT<<<cdiv(HID * QD, 256), 256, 0, stream>>>(wq, Wqt, HID, QD);
  k_cvtT<<<cdiv(HID * KD, 256), 256, 0, stream>>>(wk, Wkt, HID, KD);
  k_cvtT<<<cdiv(HID * KD, 256), 256, 0, stream>>>(wv, Wvt, HID, KD);
  k_cvtT<<<cdiv(QD * HID, 256), 256, 0, stream>>>(wo, Wot, QD, HID);

  // 2) QKV projections (LDS-tiled, TDM double-buffered bf16 WMMA)
  k_gemm_bf16<<<(TT / GBM) * (QD / GBN), 256, 0, stream>>>(Xbf, Wqt, Qf, TT, QD, HID);
  k_gemm_bf16<<<(TT / GBM) * (KD / GBN), 256, 0, stream>>>(Xbf, Wkt, Kf, TT, KD, HID);
  k_gemm_bf16<<<(TT / GBM) * (KD / GBN), 256, 0, stream>>>(Xbf, Wvt, Vf, TT, KD, HID);

  // 3) RoPE (f32 -> bf16) and V transpose (for contiguous P@V B-fragments)
  {
    int total = TT * (NH + NKV) * (HDIM / 2);     // 2359296
    k_rope<<<cdiv(total, 256), 256, 0, stream>>>(Qf, Kf, pos, Qbf, Kbf);
  }
  k_cvtT<<<cdiv(TT * KD, 256), 256, 0, stream>>>(Vf, (unsigned short*)Vtb, TT, KD);

  // 4) sliding-window attention with sinks (WMMA QK^T and PV)
  k_attn<<<(NH * (TT / 16)) / 4, 128, 0, stream>>>(Qbf, Kbf, Vtb, sinks, Attn);

  // 5) output projection -> f32 result
  k_gemm_bf16<<<(TT / GBM) * (HID / GBN), 256, 0, stream>>>(Attn, Wot, out, TT, HID, QD);
}
